// SpatialAttention_53721450939061
// MI455X (gfx1250) — compile-verified, matrix operand order checked by reading, not run on hardware
//
#include <hip/hip_runtime.h>

typedef __bf16 bf16_t;
typedef __attribute__((ext_vector_type(16))) __bf16 v16bf;
typedef __attribute__((ext_vector_type(8)))  __bf16 v8bf;
typedef __attribute__((ext_vector_type(8)))  float  v8f;
typedef __attribute__((ext_vector_type(4)))  unsigned int v4u;
typedef __attribute__((ext_vector_type(8)))  int v8i;
typedef __attribute__((ext_vector_type(4)))  int v4i;

#if defined(__has_builtin)
#if __has_builtin(__builtin_amdgcn_tensor_load_to_lds) && \
    __has_builtin(__builtin_amdgcn_s_wait_tensorcnt)
#define USE_TDM 1
#endif
#endif

// ---------------------------------------------------------------------------
// fp32 -> bf16 conversion
// ---------------------------------------------------------------------------
__global__ __launch_bounds__(256) void cvt_f32_bf16(const float* __restrict__ in,
                                                    bf16_t* __restrict__ out,
                                                    long n) {
  long i = (long)blockIdx.x * blockDim.x + threadIdx.x;
  if (i < n) out[i] = (bf16_t)in[i];
}

#ifdef USE_TDM
// ---------------------------------------------------------------------------
// Issue a TDM DMA of a 64-row x 32-element bf16 tile (row stride = K elements)
// from global memory into LDS at byte offset lds_off. Uniform args -> SGPRs.
// D# layout per CDNA5 ISA ch.8 (group0 128b, group1 256b; groups 2/3 and the
// trailing 256-bit group zero => 2-D tile, no gather, no multicast).
// This toolchain exposes the 6-argument builtin
//   (uint32x4, int32x8, int32x4, int32x4, int32x8, i32 cpol).
// ---------------------------------------------------------------------------
__device__ __forceinline__ void tdm_load_tile_64x32_bf16(const bf16_t* gptr,
                                                         unsigned lds_off,
                                                         int K) {
  const unsigned long long ga = (unsigned long long)(size_t)gptr;
  v4u g0;
  g0[0] = 1u;                                   // count=1 (valid), rest of flags 0
  g0[1] = lds_off;                              // lds_addr (bytes)
  g0[2] = (unsigned)(ga & 0xffffffffu);         // global_addr[31:0]
  g0[3] = (unsigned)((ga >> 32) & 0x1ffffffu)   // global_addr[56:32]
          | (2u << 30);                         // type = 2 ("image")
  v8i g1;
  g1[0] = (int)(1u << 16);                      // data_size=1 (2 bytes/elem)
  g1[1] = (int)(((unsigned)K & 0xffffu) << 16); // tensor_dim0[15:0]  @ bits 63:48
  g1[2] = (int)((((unsigned)K >> 16) & 0xffffu) // tensor_dim0[31:16] @ bits 79:64
          | (64u << 16));                       // tensor_dim1[15:0]=64 @ bits 95:80
  g1[3] = (int)(32u << 16);                     // tile_dim0=32 @ bits 127:112
  g1[4] = (int)64u;                             // tile_dim1=64 @ bits 143:128
  g1[5] = (int)(unsigned)K;                     // tensor_dim0_stride[31:0]
  g1[6] = 0;                                    // stride[47:32]=0, dim1_stride lo=0
  g1[7] = 0;
  v4i gz4 = {0, 0, 0, 0};
  v8i gz8 = {0, 0, 0, 0, 0, 0, 0, 0};
  __builtin_amdgcn_tensor_load_to_lds(g0, g1, gz4, gz4, gz8, 0);
}
#endif

// ---------------------------------------------------------------------------
// Generic NT GEMM:  C[M,Ncols] = scale * (A[M,K] @ B[Ncols,K]^T) (+ bias[col])
// A, B bf16 row-major. C fp32 or bf16, row-major (ldc = Ncols).
// blockIdx.z = batch; strideA/strideB/strideC per-batch element strides
// (0 => matrix shared across batch).
//
// Block = 256 threads = 8 waves; block tile 256(M) x 64(N); each wave owns a
// 32x64 strip (2 M-subtiles x 4 N-subtiles), K-steps of 32 via
// v_wmma_f32_16x16x32_bf16, fp32 accumulation.
// B strip (64x32 bf16, 4 KB) is staged into LDS by the Tensor Data Mover,
// double-buffered and tracked with TENSORcnt; A comes straight from global.
// ---------------------------------------------------------------------------
template<bool OUT_BF16, bool ADD_BIAS>
__global__ __launch_bounds__(256) void gemm_nt_bf16(
    const bf16_t* __restrict__ A, const bf16_t* __restrict__ B,
    void* __restrict__ C, const float* __restrict__ bias,
    int M, int Ncols, int K, float scale,
    long strideA, long strideB, long strideC)
{
  const int tid  = threadIdx.x;
  const int wave = tid >> 5;
  const int lane = tid & 31;
  const int r16  = lane & 15;   // row-within-16 for A/B fragment loads
  const int hi   = lane >> 4;   // half-wave selector (K-offset 0 or 8)

  const int m0 = blockIdx.y * 256 + wave * 32;
  const int n0 = blockIdx.x * 64;

  const bf16_t* Ab = A + (long)blockIdx.z * strideA;
  const bf16_t* Bb = B + (long)blockIdx.z * strideB;

  v8f acc[2][4] = {};

#ifdef USE_TDM
  __shared__ __align__(128) bf16_t Bsh[2][64 * 32];
  const unsigned lds_base = (unsigned)(size_t)(&Bsh[0][0]);
  const int nsteps = K >> 5;

  if (wave == 0)
    tdm_load_tile_64x32_bf16(Bb + (long)n0 * K, lds_base, K);

  for (int i = 0; i < nsteps; ++i) {
    const int k0 = i << 5;
    if (wave == 0) {
      if (i + 1 < nsteps) {
        tdm_load_tile_64x32_bf16(Bb + (long)n0 * K + (k0 + 32),
                                 lds_base + (unsigned)(((i + 1) & 1) * 4096), K);
        __builtin_amdgcn_s_wait_tensorcnt(1);   // TDM(i) complete
      } else {
        __builtin_amdgcn_s_wait_tensorcnt(0);
      }
    }
    __syncthreads();                            // publish Bsh[i&1] to all waves

    // A fragments from global (ISA 16-bit A layout: hi=0 -> K 0..7/16..23,
    // hi=1 -> K 8..15/24..31)
    v16bf afrag[2];
    #pragma unroll
    for (int mi = 0; mi < 2; ++mi) {
      const bf16_t* p = Ab + (long)(m0 + mi * 16 + r16) * K + k0 + hi * 8;
      v8bf lo  = *(const v8bf*)(p);
      v8bf hi8 = *(const v8bf*)(p + 16);
      if (k0 + 32 < K) __builtin_prefetch(p + 32, 0, 1);
      afrag[mi] = __builtin_shufflevector(lo, hi8,
          0, 1, 2, 3, 4, 5, 6, 7, 8, 9, 10, 11, 12, 13, 14, 15);
    }
    // B fragments from LDS tile (row r = n-within-strip, 32 elems per row)
    const bf16_t* Bbuf = &Bsh[i & 1][0];
    v16bf bfrag[4];
    #pragma unroll
    for (int ni = 0; ni < 4; ++ni) {
      const bf16_t* p = Bbuf + (ni * 16 + r16) * 32 + hi * 8;
      v8bf lo  = *(const v8bf*)(p);
      v8bf hi8 = *(const v8bf*)(p + 16);
      bfrag[ni] = __builtin_shufflevector(lo, hi8,
          0, 1, 2, 3, 4, 5, 6, 7, 8, 9, 10, 11, 12, 13, 14, 15);
    }
    #pragma unroll
    for (int mi = 0; mi < 2; ++mi)
      #pragma unroll
      for (int ni = 0; ni < 4; ++ni)
        acc[mi][ni] = __builtin_amdgcn_wmma_f32_16x16x32_bf16(
            false, afrag[mi], false, bfrag[ni],
            (short)0, acc[mi][ni], false, false);

    __syncthreads();                            // done reading Bsh[i&1]
  }
#else
  for (int k0 = 0; k0 < K; k0 += 32) {
    v16bf afrag[2];
    #pragma unroll
    for (int mi = 0; mi < 2; ++mi) {
      const bf16_t* p = Ab + (long)(m0 + mi * 16 + r16) * K + k0 + hi * 8;
      v8bf lo  = *(const v8bf*)(p);
      v8bf hi8 = *(const v8bf*)(p + 16);
      afrag[mi] = __builtin_shufflevector(lo, hi8,
          0, 1, 2, 3, 4, 5, 6, 7, 8, 9, 10, 11, 12, 13, 14, 15);
    }
    v16bf bfrag[4];
    #pragma unroll
    for (int ni = 0; ni < 4; ++ni) {
      const bf16_t* p = Bb + (long)(n0 + ni * 16 + r16) * K + k0 + hi * 8;
      v8bf lo  = *(const v8bf*)(p);
      v8bf hi8 = *(const v8bf*)(p + 16);
      bfrag[ni] = __builtin_shufflevector(lo, hi8,
          0, 1, 2, 3, 4, 5, 6, 7, 8, 9, 10, 11, 12, 13, 14, 15);
    }
    #pragma unroll
    for (int mi = 0; mi < 2; ++mi)
      #pragma unroll
      for (int ni = 0; ni < 4; ++ni)
        acc[mi][ni] = __builtin_amdgcn_wmma_f32_16x16x32_bf16(
            false, afrag[mi], false, bfrag[ni],
            (short)0, acc[mi][ni], false, false);
  }
#endif

  // Epilogue. C layout: lanes 0-15 -> M = vgpr r, lanes 16-31 -> M = r + 8,
  // N = lane % 16.
  #pragma unroll
  for (int mi = 0; mi < 2; ++mi) {
    #pragma unroll
    for (int ni = 0; ni < 4; ++ni) {
      const int col = n0 + ni * 16 + r16;
      const float badd = ADD_BIAS ? bias[col] : 0.0f;
      #pragma unroll
      for (int r = 0; r < 8; ++r) {
        const int row = m0 + mi * 16 + hi * 8 + r;
        const float v = acc[mi][ni][r] * scale + badd;
        const long idx = (long)blockIdx.z * strideC + (long)row * Ncols + col;
        if (OUT_BF16) ((bf16_t*)C)[idx] = (bf16_t)v;
        else          ((float*)C)[idx]  = v;
      }
    }
  }
}

// ---------------------------------------------------------------------------
// Row softmax over 512 columns, fp32 in -> normalized bf16 out.
// One 256-thread block per row.
// ---------------------------------------------------------------------------
__global__ __launch_bounds__(256) void softmax_rows_512(const float* __restrict__ in,
                                                        bf16_t* __restrict__ out) {
  __shared__ float red[256];
  const int tid = threadIdx.x;
  const long row = blockIdx.x;
  const float* p = in + row * 512;
  float x0 = p[tid];
  float x1 = p[tid + 256];
  float m = fmaxf(x0, x1);
  red[tid] = m;
  __syncthreads();
  for (int s = 128; s > 0; s >>= 1) {
    if (tid < s) red[tid] = fmaxf(red[tid], red[tid + s]);
    __syncthreads();
  }
  m = red[0];
  __syncthreads();
  float e0 = __expf(x0 - m);
  float e1 = __expf(x1 - m);
  red[tid] = e0 + e1;
  __syncthreads();
  for (int s = 128; s > 0; s >>= 1) {
    if (tid < s) red[tid] += red[tid + s];
    __syncthreads();
  }
  const float inv = 1.0f / red[0];
  bf16_t* q = out + row * 512;
  q[tid]       = (bf16_t)(e0 * inv);
  q[tid + 256] = (bf16_t)(e1 * inv);
}

// ---------------------------------------------------------------------------
// Launch
// ---------------------------------------------------------------------------
extern "C" void kernel_launch(void* const* d_in, const int* in_sizes, int n_in,
                              void* d_out, int out_size, void* d_ws, size_t ws_size,
                              hipStream_t stream) {
  constexpr int B = 16, T = 1024, N = 512;
  constexpr float ATT_SCALE = 0.0441941738241592f;  // 512^-0.5

  const float* x  = (const float*)d_in[0];
  const float* Wq = (const float*)d_in[1];
  const float* Wk = (const float*)d_in[2];
  const float* Wv = (const float*)d_in[3];
  const float* Wp = (const float*)d_in[4];
  const float* bp = (const float*)d_in[5];
  float* out = (float*)d_out;

  // Workspace carve:
  //   [0,16M)   x_bf   [B][T][N]
  //   [16,32M)  q_bf   [B][N][T]   (reused as px_bf [B][T][N] after scores)
  //   [32,48M)  k_bf   [B][N][T]   (reused as attn_bf [B][N][N] after scores)
  //   [48,64M)  v_bf   [B][T][N]
  //   [64,80M)  attn_f32 [B][N][N]
  //   [80,82M)  wq/wk/wv/wp bf16
  char* w = (char*)d_ws;
  bf16_t* x_bf  = (bf16_t*)(w);
  bf16_t* q_bf  = (bf16_t*)(w + (16ll << 20));
  bf16_t* k_bf  = (bf16_t*)(w + (32ll << 20));
  bf16_t* v_bf  = (bf16_t*)(w + (48ll << 20));
  float*  attnf = (float*) (w + (64ll << 20));
  bf16_t* wq_bf = (bf16_t*)(w + (80ll << 20));
  bf16_t* wk_bf = wq_bf + (long)N * N;
  bf16_t* wv_bf = wk_bf + (long)N * N;
  bf16_t* wp_bf = wv_bf + (long)N * N;
  bf16_t* attn_bf = k_bf;   // k dead after score GEMM
  bf16_t* px_bf   = q_bf;   // q dead after score GEMM

  const long nx = (long)B * T * N;
  const long nw = (long)N * N;
  cvt_f32_bf16<<<(unsigned)(nx / 256), 256, 0, stream>>>(x,  x_bf,  nx);
  cvt_f32_bf16<<<(unsigned)(nw / 256), 256, 0, stream>>>(Wq, wq_bf, nw);
  cvt_f32_bf16<<<(unsigned)(nw / 256), 256, 0, stream>>>(Wk, wk_bf, nw);
  cvt_f32_bf16<<<(unsigned)(nw / 256), 256, 0, stream>>>(Wv, wv_bf, nw);
  cvt_f32_bf16<<<(unsigned)(nw / 256), 256, 0, stream>>>(Wp, wp_bf, nw);

  const dim3 blk(256);
  const long sXT = (long)T * N;
  const long sNN = (long)N * N;

  // q[b] [N,T] = Wq @ x[b]^T
  gemm_nt_bf16<true, false><<<dim3(T / 64, N / 256, B), blk, 0, stream>>>(
      wq_bf, x_bf, q_bf, nullptr, N, T, N, 1.0f, 0, sXT, sXT);
  // k[b] [N,T]
  gemm_nt_bf16<true, false><<<dim3(T / 64, N / 256, B), blk, 0, stream>>>(
      wk_bf, x_bf, k_bf, nullptr, N, T, N, 1.0f, 0, sXT, sXT);
  // v[b] [T,N] = x[b] @ Wv^T
  gemm_nt_bf16<true, false><<<dim3(N / 64, T / 256, B), blk, 0, stream>>>(
      x_bf, wv_bf, v_bf, nullptr, T, N, N, 1.0f, sXT, 0, sXT);
  // attn[b] [N,N] = scale * q[b] @ k[b]^T  (K = T), fp32 out
  gemm_nt_bf16<false, false><<<dim3(N / 64, N / 256, B), blk, 0, stream>>>(
      q_bf, k_bf, attnf, nullptr, N, N, T, ATT_SCALE, sXT, sXT, sNN);
  // softmax rows -> bf16
  softmax_rows_512<<<(unsigned)(B * N), 256, 0, stream>>>(attnf, attn_bf);
  // px[b] [T,N] = v[b] @ attn[b]^T
  gemm_nt_bf16<true, false><<<dim3(N / 64, T / 256, B), blk, 0, stream>>>(
      v_bf, attn_bf, px_bf, nullptr, T, N, N, 1.0f, sXT, sNN, sXT);
  // out[b] [T,N] = px[b] @ Wp^T + bp  (fp32 out)
  gemm_nt_bf16<false, true><<<dim3(N / 64, T / 256, B), blk, 0, stream>>>(
      px_bf, wp_bf, out, bp, T, N, N, 1.0f, sXT, 0, sXT);
}
